// TVConv_37331855737250
// MI455X (gfx1250) — matrix operand order repaired and checked by reading, not compiled, not run on hardware
//
#include <hip/hip_runtime.h>
#include <hip/hip_bf16.h>

typedef __attribute__((ext_vector_type(16))) _Float16 v16h;
typedef __attribute__((ext_vector_type(8)))  float    v8f;

#define Hd 128
#define Wd 128
#define NPIX (Hd*Wd)
#define HP 130
#define WP 130
#define Cc 64

// ---------------- workspace layout ----------------
__host__ __device__ constexpr size_t alignup(size_t v){ return (v + 255) & ~size_t(255); }
constexpr size_t OFF_STATS = 0;                                          // 32 floats: [0..7]=sum/ss x4, [16..23]=mean/rstd x4
constexpr size_t OFF_POSI  = 256;                                        // HP*WP*4 fp32, channels-last padded
constexpr size_t OFF_Y     = alignup(OFF_POSI + (size_t)HP*WP*4*4);      // 64*NPIX fp32 (pre-norm conv out)
constexpr size_t OFF_ACT   = alignup(OFF_Y    + (size_t)Cc*NPIX*4);      // HP*WP*64 f16, channels-last padded
constexpr size_t OFF_PW1   = alignup(OFF_ACT  + (size_t)HP*WP*Cc*2);     // 4*9216 f16 packed A
constexpr size_t OFF_PW2   = alignup(OFF_PW1  + (size_t)4*9216*2);
constexpr size_t OFF_PW3   = alignup(OFF_PW2  + (size_t)4*9216*2);
constexpr size_t OFF_PWF   = alignup(OFF_PW3  + (size_t)4*9216*2);       // 36*9216 f16
constexpr size_t OFF_WMAP  = alignup(OFF_PWF  + (size_t)36*9216*2);      // 576*NPIX fp32

// ---------------- kernels ----------------
__global__ void k_zero_stats(float* stats){
    if (threadIdx.x < 32) stats[threadIdx.x] = 0.f;
}

// bilinear resize 8x8 -> 128x128 (align_corners=False), write zero-padded channels-last [HP][WP][4]
__global__ void k_resize(const float* __restrict__ pm, float* __restrict__ posi){
    int t = blockIdx.x*blockDim.x + threadIdx.x;
    if (t >= HP*WP) return;
    float v[4] = {0.f,0.f,0.f,0.f};
    int r = t / WP, col = t % WP;
    if (r >= 1 && r <= Hd && col >= 1 && col <= Wd){
        float sy = ((float)(r-1) + 0.5f)*(8.0f/128.0f) - 0.5f;
        float sx = ((float)(col-1)+ 0.5f)*(8.0f/128.0f) - 0.5f;
        float fy = floorf(sy), fx = floorf(sx);
        float wy = sy - fy,    wx = sx - fx;
        int y0 = (int)fy, x0 = (int)fx;
        int y1 = min(max(y0+1,0),7), x1 = min(max(x0+1,0),7);
        y0 = min(max(y0,0),7); x0 = min(max(x0,0),7);
        #pragma unroll
        for (int ch = 0; ch < 4; ++ch){
            const float* p = pm + ch*64;
            float a = p[y0*8+x0], b = p[y0*8+x1], c = p[y1*8+x0], d = p[y1*8+x1];
            float top = a + wx*(b-a), bot = c + wx*(d-c);
            v[ch] = top + wy*(bot-top);
        }
    }
    float* dst = posi + (size_t)t*4;
    dst[0]=v[0]; dst[1]=v[1]; dst[2]=v[2]; dst[3]=v[3];
}

// direct conv 4->64 (K=36, too small for WMMA) + GN partial stats (slot 0)
__global__ void k_conv0(const float* __restrict__ posi, const float* __restrict__ w0,
                        float* __restrict__ y, float* stats){
    __shared__ float sw[64*36];            // [co][tap*4+ci]
    __shared__ float rs[128], rss[128];
    for (int e = threadIdx.x; e < 64*36; e += blockDim.x){
        int co = e/36, rem = e%36, tap = rem>>2, ci = rem&3;
        sw[e] = w0[(co*4 + ci)*9 + tap];   // w0: [64][4][3][3]
    }
    __syncthreads();
    int x = threadIdx.x;                   // 128 threads = one row
    int yy = blockIdx.x;
    float pv[36];
    #pragma unroll
    for (int tap = 0; tap < 9; ++tap){
        int dy = tap/3, dx = tap%3;
        const float* p = posi + ((size_t)(yy+dy)*WP + (x+dx))*4;
        pv[tap*4+0]=p[0]; pv[tap*4+1]=p[1]; pv[tap*4+2]=p[2]; pv[tap*4+3]=p[3];
    }
    float s = 0.f, ss = 0.f;
    for (int co = 0; co < 64; ++co){
        float acc = 0.f;
        #pragma unroll
        for (int k = 0; k < 36; ++k) acc += sw[co*36+k]*pv[k];
        y[(size_t)co*NPIX + yy*Wd + x] = acc;
        s += acc; ss += acc*acc;
    }
    rs[threadIdx.x] = s; rss[threadIdx.x] = ss;
    __syncthreads();
    for (int off = 64; off > 0; off >>= 1){
        if (threadIdx.x < off){ rs[threadIdx.x]+=rs[threadIdx.x+off]; rss[threadIdx.x]+=rss[threadIdx.x+off]; }
        __syncthreads();
    }
    if (threadIdx.x == 0){ atomicAdd(&stats[0], rs[0]); atomicAdd(&stats[1], rss[0]); }
}

__global__ void k_finalize(float* stats, int slot){
    if (threadIdx.x == 0){
        const float n = 64.0f*128.0f*128.0f;
        float mean = stats[slot*2] / n;
        float var  = stats[slot*2+1] / n - mean*mean;
        stats[16 + slot*2]     = mean;
        stats[16 + slot*2 + 1] = rsqrtf(var + 1e-5f);
    }
}

// GN normalize + ReLU + convert to f16 channels-last zero-padded [HP][WP][64]
__global__ void k_prepare(const float* __restrict__ y, const float* __restrict__ g,
                          const float* __restrict__ b, const float* __restrict__ stats,
                          int slot, _Float16* __restrict__ act){
    int t = blockIdx.x*blockDim.x + threadIdx.x;
    if (t >= HP*WP) return;
    int r = t / WP, col = t % WP;
    unsigned* dst = (unsigned*)(act + (size_t)t*Cc);
    if (r < 1 || r > Hd || col < 1 || col > Wd){
        #pragma unroll
        for (int i = 0; i < 32; ++i) dst[i] = 0u;
        return;
    }
    float mean = stats[16 + slot*2], rstd = stats[16 + slot*2 + 1];
    const float* yp = y + (size_t)(r-1)*Wd + (col-1);
    #pragma unroll
    for (int c2 = 0; c2 < 32; ++c2){
        float v0 = yp[(size_t)(2*c2  )*NPIX];
        float v1 = yp[(size_t)(2*c2+1)*NPIX];
        v0 = fmaxf(0.f, (v0-mean)*rstd*g[2*c2  ] + b[2*c2  ]);
        v1 = fmaxf(0.f, (v1-mean)*rstd*g[2*c2+1] + b[2*c2+1]);
        union { _Float16 h[2]; unsigned u; } pk;
        pk.h[0] = (_Float16)v0; pk.h[1] = (_Float16)v1;
        dst[c2] = pk.u;
    }
}

// repack OIHW fp32 weights into a WMMA A-fragment layout (contiguous k order):
// packed[(((mt*9+tap)*2+kc)*32 + L)*16 + i] ; K(L,i) = 16*(L/16) + i (contiguous, matches B)
__global__ void k_repack(const float* __restrict__ wsrc, _Float16* __restrict__ packed, int mTiles){
    int e = blockIdx.x*blockDim.x + threadIdx.x;
    int total = mTiles*9216;
    if (e >= total) return;
    int i   =  e        & 15;
    int L   = (e >> 4)  & 31;
    int kc  = (e >> 9)  &  1;
    int tap = (e / 1024) % 9;
    int mt  =  e / 9216;
    int m   = mt*16 + (L & 15);
    int K   = (L < 16 ? 0 : 16) + i;
    int cin = kc*32 + K;
    int ky  = tap/3, kx = tap%3;
    packed[e] = (_Float16)wsrc[(((size_t)m*64 + cin)*3 + ky)*3 + kx];
}

// implicit-GEMM 3x3 conv via WMMA.
// Block: 256 threads (8 waves), one output row; 3 input rows staged in LDS
// (contiguous 49.9KB slab of the padded channels-last f16 activation buffer,
// copied with the gfx1250 async global->LDS path when available).
// Each wave: 16 pixels x (MT*16) output channels; B fragment loaded once from
// LDS (ds_load_b128) and reused across MT WMMAs -> 9*2*MT wmma per wave.
template<int MT>
__global__ void k_conv_wmma(const _Float16* __restrict__ act,
                            const _Float16* __restrict__ packedA,
                            float* __restrict__ yout,
                            float* stats, int slot){
    __shared__ _Float16 sact[3*WP*Cc];     // 49,920 B
    const int row = blockIdx.x;            // output row
    const _Float16* src = act + (size_t)row*WP*Cc;   // rows row..row+2 are contiguous
    constexpr int NCHUNK = (3*WP*Cc*2)/16; // 3120 x 16B

#if __has_builtin(__builtin_amdgcn_global_load_async_to_lds_b128)
    {
        typedef int v4i_ __attribute__((vector_size(16)));
        typedef __attribute__((address_space(1))) v4i_ GV4;
        typedef __attribute__((address_space(3))) v4i_ LV4;
        GV4* g = (GV4*)src;
        LV4* l = (LV4*)sact;
        for (int e = threadIdx.x; e < NCHUNK; e += 256)
            __builtin_amdgcn_global_load_async_to_lds_b128(g + e, l + e, 0, 0);
        asm volatile("s_wait_asynccnt 0" ::: "memory");
    }
#else
    {
        const uint4* g4 = (const uint4*)src;
        uint4* s4 = (uint4*)sact;
        for (int e = threadIdx.x; e < NCHUNK; e += 256) s4[e] = g4[e];
    }
#endif
    __syncthreads();

    const int lane = threadIdx.x & 31;
    const int wave = threadIdx.x >> 5;
    const int mt0  = blockIdx.y * MT;      // first out-channel tile
    const int x0   = wave*16;              // 8 waves cover the 128-wide row
    const int nloc = lane & 15;            // N = pixel within tile
    const int hi   = (lane >> 4) & 1;

    v8f acc[MT];
    #pragma unroll
    for (int m = 0; m < MT; ++m) acc[m] = (v8f){};

    const _Float16* aBase = packedA + (size_t)mt0*9216 + lane*16;
    #pragma unroll
    for (int tap = 0; tap < 9; ++tap){
        const int dy = tap/3, dx = tap%3;
        const _Float16* bp = sact + ((dy*WP) + x0 + nloc + dx)*Cc + hi*16;
        #pragma unroll
        for (int kc = 0; kc < 2; ++kc){
            v16h b = *(const v16h*)(bp + kc*32);
            #pragma unroll
            for (int m = 0; m < MT; ++m){
                v16h a = *(const v16h*)(aBase + (size_t)m*9216 + (tap*2 + kc)*512);
                acc[m] = __builtin_amdgcn_wmma_f32_16x16x32_f16(
                    false, a, false, b, (short)0, acc[m], false, false);
            }
        }
    }

    float s = 0.f, ss = 0.f;
    #pragma unroll
    for (int m = 0; m < MT; ++m){
        float* outp = yout + (size_t)((mt0+m)*16 + hi*8)*NPIX + (size_t)row*Wd + x0 + nloc;
        #pragma unroll
        for (int v = 0; v < 8; ++v){
            float val = acc[m][v];
            outp[(size_t)v*NPIX] = val;    // M = v + hi*8 per C/D layout
            s += val; ss += val*val;
        }
    }
    if (slot >= 0){
        __shared__ float red[512];
        red[threadIdx.x] = s; red[256 + threadIdx.x] = ss;
        __syncthreads();
        for (int off = 128; off > 0; off >>= 1){
            if (threadIdx.x < off){
                red[threadIdx.x]       += red[threadIdx.x + off];
                red[256 + threadIdx.x] += red[256 + threadIdx.x + off];
            }
            __syncthreads();
        }
        if (threadIdx.x == 0){
            atomicAdd(&stats[slot*2],     red[0]);
            atomicAdd(&stats[slot*2 + 1], red[256]);
        }
    }
}

// per-pixel dynamic 3x3 apply: weights held in registers, 8 batches looped inside
__global__ void k_apply(const float* __restrict__ x, const float* __restrict__ wmap,
                        float* __restrict__ out){
    int t = blockIdx.x*blockDim.x + threadIdx.x;     // c*NPIX + p
    if (t >= Cc*NPIX) return;
    int c  = t >> 14;
    int p  = t & (NPIX-1);
    int yy = p >> 7, xx = p & 127;
    float wk[9];
    #pragma unroll
    for (int k = 0; k < 9; ++k) wk[k] = wmap[(size_t)(c*9 + k)*NPIX + p];
    #pragma unroll
    for (int b = 0; b < 8; ++b){
        const float* xb = x + (size_t)(b*Cc + c)*NPIX;
        float acc = 0.f;
        #pragma unroll
        for (int k = 0; k < 9; ++k){
            int ry = yy + k/3 - 1, rx = xx + k%3 - 1;
            float xv = (ry >= 0 && ry < Hd && rx >= 0 && rx < Wd) ? xb[ry*Wd + rx] : 0.f;
            acc += wk[k]*xv;
        }
        out[(size_t)(b*Cc + c)*NPIX + p] = acc;
    }
}

// ---------------- launch ----------------
extern "C" void kernel_launch(void* const* d_in, const int* in_sizes, int n_in,
                              void* d_out, int out_size, void* d_ws, size_t ws_size,
                              hipStream_t stream) {
    const float* x  = (const float*)d_in[0];
    const float* pm = (const float*)d_in[1];
    const float* w0 = (const float*)d_in[2];
    const float* g0 = (const float*)d_in[3];  const float* b0 = (const float*)d_in[4];
    const float* w1 = (const float*)d_in[5];
    const float* g1 = (const float*)d_in[6];  const float* b1 = (const float*)d_in[7];
    const float* w2 = (const float*)d_in[8];
    const float* g2 = (const float*)d_in[9];  const float* b2 = (const float*)d_in[10];
    const float* w3 = (const float*)d_in[11];
    const float* g3 = (const float*)d_in[12]; const float* b3 = (const float*)d_in[13];
    const float* wf = (const float*)d_in[14];
    float* out = (float*)d_out;

    char* ws = (char*)d_ws;
    float*    stats = (float*)   (ws + OFF_STATS);
    float*    posi  = (float*)   (ws + OFF_POSI);
    float*    ybuf  = (float*)   (ws + OFF_Y);
    _Float16* act   = (_Float16*)(ws + OFF_ACT);
    _Float16* pw1   = (_Float16*)(ws + OFF_PW1);
    _Float16* pw2   = (_Float16*)(ws + OFF_PW2);
    _Float16* pw3   = (_Float16*)(ws + OFF_PW3);
    _Float16* pwf   = (_Float16*)(ws + OFF_PWF);
    float*    wmap  = (float*)   (ws + OFF_WMAP);

    const int PADPIX = HP*WP;                    // 16900
    k_zero_stats<<<1, 32, 0, stream>>>(stats);
    k_resize<<<(PADPIX+255)/256, 256, 0, stream>>>(pm, posi);
    k_repack<<<(4*9216+255)/256, 256, 0, stream>>>(w1, pw1, 4);
    k_repack<<<(4*9216+255)/256, 256, 0, stream>>>(w2, pw2, 4);
    k_repack<<<(4*9216+255)/256, 256, 0, stream>>>(w3, pw3, 4);
    k_repack<<<(36*9216+255)/256, 256, 0, stream>>>(wf, pwf, 36);

    k_conv0<<<Hd, 128, 0, stream>>>(posi, w0, ybuf, stats);
    k_finalize<<<1, 1, 0, stream>>>(stats, 0);
    k_prepare<<<(PADPIX+255)/256, 256, 0, stream>>>(ybuf, g0, b0, stats, 0, act);

    k_conv_wmma<2><<<dim3(Hd, 2), 256, 0, stream>>>(act, pw1, ybuf, stats, 1);
    k_finalize<<<1, 1, 0, stream>>>(stats, 1);
    k_prepare<<<(PADPIX+255)/256, 256, 0, stream>>>(ybuf, g1, b1, stats, 1, act);

    k_conv_wmma<2><<<dim3(Hd, 2), 256, 0, stream>>>(act, pw2, ybuf, stats, 2);
    k_finalize<<<1, 1, 0, stream>>>(stats, 2);
    k_prepare<<<(PADPIX+255)/256, 256, 0, stream>>>(ybuf, g2, b2, stats, 2, act);

    k_conv_wmma<2><<<dim3(Hd, 2), 256, 0, stream>>>(act, pw3, ybuf, stats, 3);
    k_finalize<<<1, 1, 0, stream>>>(stats, 3);
    k_prepare<<<(PADPIX+255)/256, 256, 0, stream>>>(ybuf, g3, b3, stats, 3, act);

    k_conv_wmma<4><<<dim3(Hd, 9), 256, 0, stream>>>(act, pwf, wmap, stats, -1);

    k_apply<<<(Cc*NPIX)/256, 256, 0, stream>>>(x, wmap, out);
}
